// GRU_Cell_68289980006875
// MI455X (gfx1250) — matrix operand order repaired and checked by reading, not compiled, not run on hardware
//
#include <hip/hip_runtime.h>

// ---------------------------------------------------------------------------
// GRU-GCN cell for MI455X (gfx1250, wave32, WMMA bf16 16x16x32).
// Algebraic simplification: affinity is antisymmetric => adj2 == adj1 exactly,
// halving GCN matmuls and folding KSUP 5 -> 3 in the per-node weight einsum.
// Heavy GEMMs (dis layer1, affinity, adj@x propagation, xg@Wfold) run on
// v_wmma_f32_16x16x32_bf16 with manually packed operand layouts per the CDNA5
// ISA (A: row in lane, two 16B chunks; B: column per lane, two 16B chunks).
// GEMM waves hold one A tile and accumulate multiple output tiles (5x / 8x
// A-operand reuse; back-to-back WMMA chains for the XDL pipe).
// ---------------------------------------------------------------------------

#define DEV __device__ __forceinline__

typedef __attribute__((ext_vector_type(16))) __bf16 v16bf;
typedef __attribute__((ext_vector_type(8)))  float  v8f;

static constexpr int Bb   = 8;
static constexpr int Nn   = 2048;
static constexpr int BN   = Bb * Nn;       // 16384
static constexpr int Cc   = 66;            // DIN + DOUT
static constexpr int Cp   = 80;            // padded feature cols (5 tiles)
static constexpr int KIp  = 224;           // padded 3*66=198 -> 224 (7 k-steps)
static constexpr int KD   = 2112;          // N + DOUT (dis layer-1 K)

struct Ptrs { const float* p[45]; };

DEV unsigned short f2bf(float f) {
  unsigned u = __float_as_uint(f);
  unsigned r = u + 0x7fffu + ((u >> 16) & 1u);
  return (unsigned short)(r >> 16);
}
DEV float sigf(float x) {                       // v_rcp instead of IEEE div
  return __builtin_amdgcn_rcpf(1.0f + __expf(-x));
}
DEV float tanh_fast(float x) {                  // 1 - 2/(e^{2x}+1), rcp-based
  float e = __expf(2.0f * x);
  return 1.0f - 2.0f * __builtin_amdgcn_rcpf(e + 1.0f);
}

union ABv { v16bf v; uint4 q[2]; unsigned short h[16]; };
union Pk8 { uint4 q; unsigned short h[8]; };

DEV v8f wmma_bf16(v16bf a, v16bf b, v8f c) {
  return __builtin_amdgcn_wmma_f32_16x16x32_bf16(false, a, false, b, (short)0, c,
                                                 false, false);
}

// A operand: bf16 row-major matrix (ld elements per row). Lane L owns row
// (row0 + L&15); VGPRs 0-3 hold K = kb..kb+7, VGPRs 4-7 hold K = kb+16..kb+23,
// kb = (L<16) ? 0 : 8  => two contiguous 16B chunks.
DEV v16bf loadA(const unsigned short* base, int ld, int row0, int k0, int lane) {
  int m  = lane & 15;
  int kb = (lane < 16) ? 0 : 8;
  const unsigned short* p = base + (size_t)(row0 + m) * ld + k0 + kb;
  ABv a;
  a.q[0] = *(const uint4*)(p);
  a.q[1] = *(const uint4*)(p + 16);
  return a.v;
}

// B operand from K-contiguous storage: element (k, col) at base[col*ld + k].
// Lane L owns col (col0 + L&15), K = kb..kb+7 and kb+16..kb+23, kb = (L<16) ? 0 : 8
// => two contiguous 16B chunks (same K order as loadA).
DEV v16bf loadB(const unsigned short* base, int ld, int col0, int k0, int lane) {
  int c  = lane & 15;
  int kb = (lane < 16) ? 0 : 8;
  const unsigned short* p = base + (size_t)(col0 + c) * ld + k0 + kb;
  ABv a;
  a.q[0] = *(const uint4*)(p);
  a.q[1] = *(const uint4*)(p + 16);
  return a.v;
}

// D tile (f32): lane L owns col (col0 + L&15); VGPR v = row (row0 + v + (L<16?0:8)).
DEV void storeD_f32(float* base, int ld, int row0, int col0, const v8f& d, int lane) {
  int c  = lane & 15;
  int mo = (lane < 16) ? 0 : 8;
#pragma unroll
  for (int v = 0; v < 8; ++v)
    base[(size_t)(row0 + mo + v) * ld + col0 + c] = d[v];
}

// ---------------------------------------------------------------------------
// K0: convert dis w1 weights (2112x16 f32, two branches) to K-contiguous bf16.
__global__ void prep_wdis_kernel(Ptrs P, unsigned short* wdis) {
  int idx = blockIdx.x * blockDim.x + threadIdx.x;       // branch*16*2112
  if (idx >= 2 * 16 * KD) return;
  int br = idx / (16 * KD);
  int r  = idx - br * 16 * KD;
  int c  = r / KD;
  int k  = r - c * KD;
  const float* w1 = P.p[17 + br * 6];                    // dis1.w1 / dis2.w1
  wdis[idx] = f2bf(w1[(size_t)k * 16 + c]);
}

// ---------------------------------------------------------------------------
// K1: tra1/tra2 + pop1/pop2 small MLPs; m{1,2}[bn][c] = z_tra*z_pop.
__global__ void feat_kernel(Ptrs P, float* m1, float* m2) {
  int bn = blockIdx.x * blockDim.x + threadIdx.x;
  if (bn >= BN) return;
  const float* x  = P.p[0];
  const float* st = P.p[1];
  const float* pp = P.p[3];

  float in[70];
  // ---- tra branches: concat(x, state) -> 16 -> 2 -> 16
  in[0] = x[(size_t)bn * 2 + 0];
  in[1] = x[(size_t)bn * 2 + 1];
  for (int i = 0; i < 64; ++i) in[2 + i] = st[(size_t)bn * 64 + i];

  float ztra[2][16];
  for (int br = 0; br < 2; ++br) {
    const float* w1 = P.p[5 + br * 6 + 0];
    const float* b1 = P.p[5 + br * 6 + 1];
    const float* w2 = P.p[5 + br * 6 + 2];
    const float* b2 = P.p[5 + br * 6 + 3];
    const float* w3 = P.p[5 + br * 6 + 4];
    const float* b3 = P.p[5 + br * 6 + 5];
    float h[16];
    for (int j = 0; j < 16; ++j) {
      float a = b1[j];
      for (int i = 0; i < 66; ++i) a = fmaf(in[i], w1[i * 16 + j], a);
      h[j] = sigf(a);
    }
    float g[2];
#pragma unroll
    for (int j = 0; j < 2; ++j) {
      float a = b2[j];
#pragma unroll
      for (int i = 0; i < 16; ++i) a = fmaf(h[i], w2[i * 2 + j], a);
      g[j] = sigf(a);
    }
#pragma unroll
    for (int c = 0; c < 16; ++c)
      ztra[br][c] = b3[c] + g[0] * w3[c] + g[1] * w3[16 + c];
  }

  // ---- pop branches: concat(pop, state) -> 64 -> 16 -> 16
  for (int i = 0; i < 6; ++i)  in[i]     = pp[(size_t)bn * 6 + i];
  for (int i = 0; i < 64; ++i) in[6 + i] = st[(size_t)bn * 64 + i];

  for (int br = 0; br < 2; ++br) {
    const float* w1 = P.p[29 + br * 6 + 0];
    const float* b1 = P.p[29 + br * 6 + 1];
    const float* w2 = P.p[29 + br * 6 + 2];
    const float* b2 = P.p[29 + br * 6 + 3];
    const float* w3 = P.p[29 + br * 6 + 4];
    const float* b3 = P.p[29 + br * 6 + 5];
    float h[64];
    for (int j = 0; j < 64; ++j) {
      float a = b1[j];
      for (int i = 0; i < 70; ++i) a = fmaf(in[i], w1[i * 64 + j], a);
      h[j] = sigf(a);
    }
    float h2[16];
    for (int j = 0; j < 16; ++j) {
      float a = b2[j];
      for (int i = 0; i < 64; ++i) a = fmaf(h[i], w2[i * 16 + j], a);
      h2[j] = sigf(a);
    }
    float* mo = (br == 0) ? m1 : m2;
#pragma unroll
    for (int c = 0; c < 16; ++c) {
      float a = b3[c];
#pragma unroll
      for (int i = 0; i < 16; ++i) a = fmaf(h2[i], w3[i * 16 + c], a);
      mo[(size_t)bn * 16 + c] = ztra[br][c] * a;   // z_tra * z_pop (z_dis later)
    }
  }
}

// ---------------------------------------------------------------------------
// K2: dis layer-1 as bf16 WMMA GEMM (16384 x 2112 x 32), then sigmoid + the
// 16->2->16 tail, then e = tanh(m * z_dis). Writes e1/e2/(-e2) bf16 padded K32.
__global__ void dis_kernel(Ptrs P, const unsigned short* wdis,
                           const float* m1, const float* m2,
                           unsigned short* e1b, unsigned short* e2b,
                           unsigned short* e2nb) {
  __shared__ float h1lds[4][16][33];
  int lane = threadIdx.x & 31;
  int wv   = threadIdx.x >> 5;
  int tile = blockIdx.x * 4 + wv;                 // 0..1023
  int bn0  = tile * 16;
  const float* dist = P.p[2];
  const float* st   = P.p[1];

  int m    = lane & 15;
  int abas = (lane < 16) ? 0 : 8;
  int row  = bn0 + m;

  v8f acc1 = {};
  v8f acc2 = {};
  for (int k = 0; k < KD; k += 32) {
    // build A tile (f32 -> bf16): two 8-element chunks at k+abas, k+abas+16
    ABv a;
#pragma unroll
    for (int half = 0; half < 2; ++half) {
      int ks = k + abas + half * 16;
      const float* p = (ks < Nn) ? (dist + (size_t)row * Nn + ks)
                                 : (st + (size_t)row * 64 + (ks - Nn));
      float4 q0 = *(const float4*)(p);
      float4 q1 = *(const float4*)(p + 4);
      a.h[half * 8 + 0] = f2bf(q0.x); a.h[half * 8 + 1] = f2bf(q0.y);
      a.h[half * 8 + 2] = f2bf(q0.z); a.h[half * 8 + 3] = f2bf(q0.w);
      a.h[half * 8 + 4] = f2bf(q1.x); a.h[half * 8 + 5] = f2bf(q1.y);
      a.h[half * 8 + 6] = f2bf(q1.z); a.h[half * 8 + 7] = f2bf(q1.w);
    }
    v16bf b1 = loadB(wdis, KD, 0, k, lane);
    v16bf b2 = loadB(wdis + 16 * KD, KD, 0, k, lane);
    acc1 = wmma_bf16(a.v, b1, acc1);
    acc2 = wmma_bf16(a.v, b2, acc2);
  }

  // sigmoid(acc + b1) -> LDS, striped per the D layout
  int c  = lane & 15;
  int mo = (lane < 16) ? 0 : 8;
  const float* db1 = P.p[18];
  const float* db2 = P.p[24];
#pragma unroll
  for (int v = 0; v < 8; ++v) {
    h1lds[wv][mo + v][c]      = sigf(acc1[v] + db1[c]);
    h1lds[wv][mo + v][16 + c] = sigf(acc2[v] + db2[c]);
  }
  __syncthreads();

  // tail: lane half (0->dis1/e1, 1->dis2/e2), row = lane&15
  int half = lane >> 4;
  int r    = lane & 15;
  int bn   = bn0 + r;
  int base = 17 + half * 6;
  const float* w2 = P.p[base + 2];
  const float* b2 = P.p[base + 3];
  const float* w3 = P.p[base + 4];
  const float* b3 = P.p[base + 5];

  float h1[16];
#pragma unroll
  for (int i = 0; i < 16; ++i) h1[i] = h1lds[wv][r][half * 16 + i];
  float g[2];
#pragma unroll
  for (int j = 0; j < 2; ++j) {
    float a = b2[j];
#pragma unroll
    for (int i = 0; i < 16; ++i) a = fmaf(h1[i], w2[i * 2 + j], a);
    g[j] = sigf(a);
  }
  const float* mm = (half == 0) ? m1 : m2;
#pragma unroll
  for (int cc = 0; cc < 16; ++cc) {
    float z = b3[cc] + g[0] * w3[cc] + g[1] * w3[16 + cc];
    float e = tanh_fast(mm[(size_t)bn * 16 + cc] * z);
    unsigned short eb = f2bf(e);
    if (half == 0) {
      e1b[(size_t)bn * 32 + cc] = eb;      e1b[(size_t)bn * 32 + 16 + cc] = 0;
    } else {
      e2b[(size_t)bn * 32 + cc]  = eb;     e2b[(size_t)bn * 32 + 16 + cc]  = 0;
      e2nb[(size_t)bn * 32 + cc] = f2bf(-e); e2nb[(size_t)bn * 32 + 16 + cc] = 0;
    }
  }
}

// ---------------------------------------------------------------------------
// K3: aff_tile = relu(e1_I e2_J^T - e2_I e1_J^T), two WMMAs per 16x16 tile
// (second term via pre-negated e2). K=32 with zero padding above 16.
__global__ void affinity_kernel(const unsigned short* e1b,
                                const unsigned short* e2b,
                                const unsigned short* e2nb, float* S) {
  int lane = threadIdx.x & 31;
  int wv   = threadIdx.x >> 5;
  int t    = blockIdx.x * 8 + wv;          // 8 * 128 * 128 tiles
  int b    = t >> 14;
  int rem  = t & 16383;
  int i0   = (rem >> 7) << 4;
  int j0   = (rem & 127) << 4;
  const unsigned short* e1 = e1b + (size_t)b * Nn * 32;
  const unsigned short* e2 = e2b + (size_t)b * Nn * 32;
  const unsigned short* en = e2nb + (size_t)b * Nn * 32;

  v16bf a1  = loadA(e1, 32, i0, 0, lane);
  v16bf a2n = loadA(en, 32, i0, 0, lane);
  v16bf bb1 = loadB(e1, 32, j0, 0, lane);
  v16bf bb2 = loadB(e2, 32, j0, 0, lane);
  v8f acc = {};
  acc = wmma_bf16(a2n, bb1, acc);          // -e2 e1^T
  acc = wmma_bf16(a1, bb2, acc);           // +e1 e2^T
#pragma unroll
  for (int v = 0; v < 8; ++v) acc[v] = fmaxf(acc[v], 0.0f);
  storeD_f32(S + (size_t)b * Nn * Nn, Nn, i0, j0, acc, lane);
}

// ---------------------------------------------------------------------------
// K4: row softmax of relu'd affinity -> adj (bf16). One block per row.
__global__ void softmax_kernel(const float* S, unsigned short* adj) {
  __shared__ float red[256];
  int t  = threadIdx.x;
  size_t bn = blockIdx.x;
  const float* row = S + bn * Nn;
  float v[8];
  float lmax = -1e30f;
#pragma unroll
  for (int j = 0; j < 8; ++j) {
    v[j] = row[t + j * 256];
    lmax = fmaxf(lmax, v[j]);
  }
  red[t] = lmax; __syncthreads();
  for (int s = 128; s > 0; s >>= 1) {
    if (t < s) red[t] = fmaxf(red[t], red[t + s]);
    __syncthreads();
  }
  float gmax = red[0]; __syncthreads();
  float lsum = 0.0f;
#pragma unroll
  for (int j = 0; j < 8; ++j) {
    v[j] = __expf(v[j] - gmax);
    lsum += v[j];
  }
  red[t] = lsum; __syncthreads();
  for (int s = 128; s > 0; s >>= 1) {
    if (t < s) red[t] += red[t + s];
    __syncthreads();
  }
  float inv = __builtin_amdgcn_rcpf(red[0]);
#pragma unroll
  for (int j = 0; j < 8; ++j)
    adj[bn * Nn + t + j * 256] = f2bf(v[j] * inv);
}

// ---------------------------------------------------------------------------
// K5: build xin in K-contiguous bf16 layout xinT[b][c][n] (c<80, zero padded)
// and xg k-slot 0 (row-major [bn][224]) + zero the 198..223 pad.
// mode 0: feat = concat(x, state); mode 1: feat = concat(x, u*state).
__global__ void prep_xin_kernel(const float* x, const float* st, const float* u,
                                int mode, unsigned short* xinT,
                                unsigned short* xg) {
  int c  = threadIdx.x;            // blockDim = 128, use c < 96
  int bn = blockIdx.x;
  int b  = bn >> 11;
  int n  = bn & 2047;
  if (c < Cp) {
    float val = 0.0f;
    if (c < 2) val = x[(size_t)bn * 2 + c];
    else if (c < Cc) {
      float s = st[(size_t)bn * 64 + (c - 2)];
      val = mode ? u[(size_t)bn * 64 + (c - 2)] * s : s;
    }
    unsigned short hb = f2bf(val);
    xinT[(size_t)b * Cp * Nn + (size_t)c * Nn + n] = hb;
    if (c < Cc) xg[(size_t)bn * KIp + c] = hb;
  }
  if (c >= 70 && c < 96) xg[(size_t)bn * KIp + 198 + (c - 70)] = 0;
}

// ---------------------------------------------------------------------------
// K6: y = adj[b] @ xsrc (2048x2048 @ 2048x80 bf16 WMMA). One wave owns one
// 16-row A stripe and accumulates all 5 feature-column tiles (5x A reuse,
// back-to-back WMMA chain per k-step). Writes ydstT and xg k-slot `kslot`.
__global__ void gcn_mm_kernel(const unsigned short* adj,
                              const unsigned short* xsrcT,
                              unsigned short* ydstT, unsigned short* xg,
                              int kslot) {
  int lane = threadIdx.x & 31;
  int wv   = threadIdx.x >> 5;
  int t    = blockIdx.x * 8 + wv;          // 8 * 128 = 1024 waves
  int b    = t >> 7;
  int i0   = (t & 127) << 4;
  const unsigned short* adj_b = adj + (size_t)b * Nn * Nn;
  const unsigned short* xs_b  = xsrcT + (size_t)b * Cp * Nn;

  v8f acc[5] = {};
  for (int k = 0; k < Nn; k += 32) {
    if (k + 128 < Nn)
      __builtin_prefetch(adj_b + (size_t)(i0 + (lane & 15)) * Nn + k + 128, 0, 0);
    v16bf a = loadA(adj_b, Nn, i0, k, lane);
#pragma unroll
    for (int ct = 0; ct < 5; ++ct) {
      v16bf bt = loadB(xs_b, Nn, ct * 16, k, lane);
      acc[ct] = wmma_bf16(a, bt, acc[ct]);
    }
  }

  int mo = (lane < 16) ? 0 : 8;
#pragma unroll
  for (int ct = 0; ct < 5; ++ct) {
    int c = ct * 16 + (lane & 15);
    Pk8 pk;
#pragma unroll
    for (int v = 0; v < 8; ++v) pk.h[v] = f2bf(acc[ct][v]);
    if (ydstT)
      *(uint4*)(ydstT + (size_t)b * Cp * Nn + (size_t)c * Nn + i0 + mo) = pk.q;
    if (c < Cc) {
#pragma unroll
      for (int v = 0; v < 8; ++v)
        xg[(size_t)(b * Nn + i0 + mo + v) * KIp + kslot * Cc + c] = pk.h[v];
    }
  }
}

// ---------------------------------------------------------------------------
// K7: fold wp (ED,5,66,O) -> Wfold_cm[(d*O+o)*224 + ki] bf16 with the
// adj1==adj2 k-fold (k1+=k3, k2+=k4) and zero K padding.
__global__ void prep_wfold_kernel(const float* wp, int O, unsigned short* wf) {
  int idx = blockIdx.x * blockDim.x + threadIdx.x;
  int total = 16 * O * KIp;
  if (idx >= total) return;
  int col = idx / KIp;
  int ki  = idx - col * KIp;
  int d   = col / O;
  int o   = col - d * O;
  float v = 0.0f;
  if (ki < 3 * Cc) {
    int k = ki / Cc;
    int i = ki - k * Cc;
    v = wp[(size_t)((d * 5 + k) * Cc + i) * O + o];
    if (k >= 1) v += wp[(size_t)((d * 5 + k + 2) * Cc + i) * O + o];
  }
  wf[(size_t)col * KIp + ki] = f2bf(v);
}

// ---------------------------------------------------------------------------
// K8: P = xg (16384x224) @ Wfold (224 x 16*O), bf16 WMMA, f32 out.
// One wave owns one 16-row A stripe and an 8-tile (128-col) group: 8x A reuse.
__global__ void pgemm_kernel(const unsigned short* xg, const unsigned short* wf,
                             int O, float* Pout) {
  int lane   = threadIdx.x & 31;
  int wv     = threadIdx.x >> 5;
  int t      = blockIdx.x * 8 + wv;
  int cols   = 16 * O;
  int groups = cols >> 7;                  // 128-col groups
  int r0     = (t / groups) << 4;
  int cg     = (t % groups) << 7;
  v8f acc[8] = {};
#pragma unroll
  for (int k = 0; k < KIp; k += 32) {
    v16bf a = loadA(xg, KIp, r0, k, lane);
#pragma unroll
    for (int j = 0; j < 8; ++j) {
      v16bf bt = loadB(wf, KIp, cg + j * 16, k, lane);
      acc[j] = wmma_bf16(a, bt, acc[j]);
    }
  }
#pragma unroll
  for (int j = 0; j < 8; ++j)
    storeD_f32(Pout, cols, r0, cg + j * 16, acc[j], lane);
}

// ---------------------------------------------------------------------------
// K9: gate epilogue: sigmoid(sum_d emb[n,d] * (P[bn, d*128+o] + bp[d,o])) -> u,r
__global__ void epi_gate_kernel(const float* Pbuf, const float* emb,
                                const float* bp, float* u, float* r) {
  int idx = blockIdx.x * blockDim.x + threadIdx.x;
  if (idx >= BN * 128) return;
  int bn = idx >> 7;
  int o  = idx & 127;
  int n  = bn & 2047;
  float s = 0.0f;
#pragma unroll
  for (int d = 0; d < 16; ++d)
    s = fmaf(emb[(size_t)n * 16 + d],
             Pbuf[(size_t)bn * 2048 + d * 128 + o] + bp[d * 128 + o], s);
  float v = sigf(s);
  if (o < 64) u[(size_t)bn * 64 + o] = v;
  else        r[(size_t)bn * 64 + (o - 64)] = v;
}

// K10: final epilogue: hc = tanh(...); h = r*state + (1-r)*hc
__global__ void epi_final_kernel(const float* Pbuf, const float* emb,
                                 const float* bp, const float* r,
                                 const float* st, float* out) {
  int idx = blockIdx.x * blockDim.x + threadIdx.x;
  if (idx >= BN * 64) return;
  int bn = idx >> 6;
  int o  = idx & 63;
  int n  = bn & 2047;
  float s = 0.0f;
#pragma unroll
  for (int d = 0; d < 16; ++d)
    s = fmaf(emb[(size_t)n * 16 + d],
             Pbuf[(size_t)bn * 1024 + d * 64 + o] + bp[d * 64 + o], s);
  float hc = tanh_fast(s);
  float rv = r[idx];
  out[idx] = rv * st[idx] + (1.0f - rv) * hc;
}

// ---------------------------------------------------------------------------
extern "C" void kernel_launch(void* const* d_in, const int* in_sizes, int n_in,
                              void* d_out, int out_size, void* d_ws,
                              size_t ws_size, hipStream_t stream) {
  (void)in_sizes; (void)n_in; (void)out_size; (void)ws_size;
  Ptrs P;
  for (int i = 0; i < 45; ++i) P.p[i] = (const float*)d_in[i];

  char* w = (char*)d_ws;
  size_t off = 0;
  auto alloc = [&](size_t bytes) -> char* {
    char* r = w + off;
    off += (bytes + 255) & ~(size_t)255;
    return r;
  };

  float*          S_P  = (float*)alloc((size_t)Bb * Nn * Nn * 4);   // S, then P
  unsigned short* adj  = (unsigned short*)alloc((size_t)Bb * Nn * Nn * 2);
  unsigned short* e1b  = (unsigned short*)alloc((size_t)BN * 32 * 2);
  unsigned short* e2b  = (unsigned short*)alloc((size_t)BN * 32 * 2);
  unsigned short* e2nb = (unsigned short*)alloc((size_t)BN * 32 * 2);
  float*          m1   = (float*)alloc((size_t)BN * 16 * 4);
  float*          m2   = (float*)alloc((size_t)BN * 16 * 4);
  unsigned short* wdis = (unsigned short*)alloc((size_t)2 * 16 * KD * 2);
  unsigned short* xinT = (unsigned short*)alloc((size_t)Bb * Cp * Nn * 2);
  unsigned short* y1T  = (unsigned short*)alloc((size_t)Bb * Cp * Nn * 2);
  unsigned short* xgG  = (unsigned short*)alloc((size_t)BN * KIp * 2);
  unsigned short* xgU  = (unsigned short*)alloc((size_t)BN * KIp * 2);
  unsigned short* wfG  = (unsigned short*)alloc((size_t)16 * 128 * KIp * 2);
  unsigned short* wfU  = (unsigned short*)alloc((size_t)16 * 64 * KIp * 2);
  float*          ubuf = (float*)alloc((size_t)BN * 64 * 4);
  float*          rbuf = (float*)alloc((size_t)BN * 64 * 4);

  // ---- feature / dis / affinity ----
  prep_wdis_kernel<<<(2 * 16 * KD + 255) / 256, 256, 0, stream>>>(P, wdis);
  feat_kernel<<<BN / 128, 128, 0, stream>>>(P, m1, m2);
  dis_kernel<<<256, 128, 0, stream>>>(P, wdis, m1, m2, e1b, e2b, e2nb);
  affinity_kernel<<<16384, 256, 0, stream>>>(e1b, e2b, e2nb, S_P);
  softmax_kernel<<<BN, 256, 0, stream>>>(S_P, adj);

  // ---- folded per-node weights ----
  prep_wfold_kernel<<<(16 * 128 * KIp + 255) / 256, 256, 0, stream>>>(
      P.p[41], 128, wfG);
  prep_wfold_kernel<<<(16 * 64 * KIp + 255) / 256, 256, 0, stream>>>(
      P.p[43], 64, wfU);

  // ---- gate GCN ----
  prep_xin_kernel<<<BN, 128, 0, stream>>>(P.p[0], P.p[1], nullptr, 0, xinT, xgG);
  gcn_mm_kernel<<<128, 256, 0, stream>>>(adj, xinT, y1T, xgG, 1);
  gcn_mm_kernel<<<128, 256, 0, stream>>>(adj, y1T, nullptr, xgG, 2);
  pgemm_kernel<<<(1024 * 16) / 8, 256, 0, stream>>>(xgG, wfG, 128, S_P);
  epi_gate_kernel<<<(BN * 128) / 256, 256, 0, stream>>>(S_P, P.p[4], P.p[42],
                                                        ubuf, rbuf);

  // ---- update GCN ----
  prep_xin_kernel<<<BN, 128, 0, stream>>>(P.p[0], P.p[1], ubuf, 1, xinT, xgU);
  gcn_mm_kernel<<<128, 256, 0, stream>>>(adj, xinT, y1T, xgU, 1);
  gcn_mm_kernel<<<128, 256, 0, stream>>>(adj, y1T, nullptr, xgU, 2);
  pgemm_kernel<<<(1024 * 8) / 8, 256, 0, stream>>>(xgU, wfU, 64, S_P);
  epi_final_kernel<<<(BN * 64) / 256, 256, 0, stream>>>(S_P, P.p[4], P.p[44],
                                                        rbuf, P.p[1],
                                                        (float*)d_out);
}
